// PointTransformerMat_42047729828491
// MI455X (gfx1250) — compile-verified, matrix operand order checked by reading, not run on hardware
//
#include <hip/hip_runtime.h>
#include <stdint.h>

typedef _Float16 half_t;
typedef __attribute__((ext_vector_type(8)))  _Float16 v8h;
typedef __attribute__((ext_vector_type(16))) _Float16 v16h;
typedef __attribute__((ext_vector_type(8)))  float    v8f;

#define B_   2
#define N_   2048
#define C_   256
#define H_   4
#define HD_  64
#define L_   4
#define J_   4096   /* B*N */
#define Z_   8      /* B*H */

#define BM 64
#define BN 128
#define BK 32
#define THREADS 256

// Async global->LDS copy (CDNA5 GLOBAL_LOAD_ASYNC_TO_LDS_B128, ASYNCcnt path).
#if defined(__has_builtin)
#if __has_builtin(__builtin_amdgcn_global_load_async_to_lds_b128)
#define HAVE_ASYNC 1
#endif
#endif
#ifndef HAVE_ASYNC
#define HAVE_ASYNC 0
#endif

#if HAVE_ASYNC
// Builtin takes (int4 addrspace(1)*, int4 addrspace(3)*, imm offset, imm cpol)
typedef int v4i_vs __attribute__((vector_size(16)));
typedef __attribute__((address_space(1))) v4i_vs as1_v4i;
typedef __attribute__((address_space(3))) v4i_vs as3_v4i;
#define ASYNC_CP_B128(gsrc, lds)                                              \
  __builtin_amdgcn_global_load_async_to_lds_b128(                             \
      (as1_v4i*)(void*)(gsrc), (as3_v4i*)(void*)(lds), 0, 0)
#endif

// ---------------------------------------------------------------------------
// Fragment gather from an LDS tile stored as [rows][stride] halves, where the
// fragment K-pairs are contiguous (A tile row-major in K; B tile stored
// transposed as [n][k]).  Layout per CDNA5 ISA 7.12.2 (16-bit A 16x32):
// lanes 0-15: M=lane, K-base 0; lanes 16-31: M=lane-16, K-base 8;
// vgpr e<4 -> K = kb + 2e, e>=4 -> K = kb + 16 + 2(e-4); pairs contiguous.
// ---------------------------------------------------------------------------
__device__ __forceinline__ v16h frag_ld(const half_t* tile, int row0, int stride, int lane) {
  union { v16h v; unsigned int u[8]; } f;
  const int r  = row0 + (lane & 15);
  const int kb = (lane < 16) ? 0 : 8;
  const char* p = (const char*)(tile + (size_t)r * stride);
#pragma unroll
  for (int e = 0; e < 8; ++e) {
    int k = kb + ((e < 4) ? (2 * e) : (16 + 2 * (e - 4)));
    f.u[e] = *(const unsigned int*)(p + 2 * k);
  }
  return f.v;
}

// ---------------------------------------------------------------------------
// Shared WMMA GEMM core.  Block tile 64(M) x 128(N), K step 32.
// 256 threads = 8 waves, wave grid 2x4, each wave owns 32x32 (2x2 frags).
// AT=false: A is (M x K), lda = row stride.  AT=true: A stored (K x M).
// B is (K x Ncols), ldb = row stride.  Epi(r, c, acc) writes results.
// ---------------------------------------------------------------------------
template <bool AT, typename Epi>
__device__ __forceinline__ void gemm_core(const half_t* __restrict__ A, int lda,
                                          const half_t* __restrict__ Bm, int ldb,
                                          int K, Epi epi) {
  __shared__ half_t As[BM][BK + 8];   // [m][k]
  __shared__ half_t Bs[BN][BK + 8];   // [n][k]  (K-transposed tile)

  const int tid  = threadIdx.x;
  const int lane = tid & 31;
  const int wave = tid >> 5;
  const int wm   = wave >> 2;   // 0..1
  const int wn   = wave & 3;    // 0..3
  const int m_block = blockIdx.y * BM;
  const int n_block = blockIdx.x * BN;

  v8f acc[2][2] = {};

  for (int k0 = 0; k0 < K; k0 += BK) {
    __syncthreads();
    // ---- stage A tile (64x32 halves) ----
    if (!AT) {
      int idx = tid * 8;                 // 0..2047
      int m = idx >> 5, k = idx & 31;
      const half_t* gsrc = A + (size_t)(m_block + m) * lda + (k0 + k);
#if HAVE_ASYNC
      // VGPR-free DMA: global -> LDS, tracked by ASYNCcnt
      ASYNC_CP_B128(gsrc, &As[m][k]);
#else
      *(v8h*)&As[m][k] = *(const v8h*)gsrc;
#endif
    } else {
      int idx = tid * 8;                 // 0..2047
      int k = idx >> 6, m = idx & 63;    // 64 m per k-row
      v8h va = *(const v8h*)(A + (size_t)(k0 + k) * lda + (m_block + m));
#pragma unroll
      for (int i = 0; i < 8; ++i) As[m + i][k] = va[i];
    }
    // ---- stage B tile (32x128 halves), transposed into Bs[n][k] ----
#pragma unroll
    for (int it = 0; it < 2; ++it) {
      int idx = (tid + it * THREADS) * 8;     // 0..4095
      int k = idx >> 7, n = idx & 127;
      v8h vb = *(const v8h*)(Bm + (size_t)(k0 + k) * ldb + (n_block + n));
#pragma unroll
      for (int i = 0; i < 8; ++i) Bs[n + i][k] = vb[i];
    }
    if (k0 + BK < K) {  // prefetch next K tile (global_prefetch_b8)
      __builtin_prefetch(Bm + (size_t)(k0 + BK) * ldb + n_block + (tid & 127), 0, 1);
    }
#if HAVE_ASYNC
    if (!AT) asm volatile("s_wait_asynccnt 0x0" ::: "memory");
#endif
    __syncthreads();

    v16h af[2], bf[2];
#pragma unroll
    for (int i = 0; i < 2; ++i) af[i] = frag_ld(&As[0][0], wm * 32 + i * 16, BK + 8, lane);
#pragma unroll
    for (int j = 0; j < 2; ++j) bf[j] = frag_ld(&Bs[0][0], wn * 32 + j * 16, BK + 8, lane);
#pragma unroll
    for (int i = 0; i < 2; ++i)
#pragma unroll
      for (int j = 0; j < 2; ++j)
        acc[i][j] = __builtin_amdgcn_wmma_f32_16x16x32_f16(
            false, af[i], false, bf[j], (short)0, acc[i][j], false, false);
  }

  // ---- epilogue: D layout = vgpr r -> row r (lanes 0-15) / 8+r (lanes 16-31),
  //      col = lane & 15 ----
#pragma unroll
  for (int i = 0; i < 2; ++i)
#pragma unroll
    for (int j = 0; j < 2; ++j) {
      int rbase = m_block + wm * 32 + i * 16 + ((lane >> 4) ? 8 : 0);
      int cidx  = n_block + wn * 32 + j * 16 + (lane & 15);
#pragma unroll
      for (int r = 0; r < 8; ++r) epi(rbase + r, cidx, acc[i][j][r]);
    }
}

// ------------------------------ epilogues ----------------------------------
struct EpiSBRelu {  // BN(scale,bias)+ReLU, dual fp32/f16 store
  float* of; half_t* oh; int ld; const float* sc; const float* bi;
  __device__ void operator()(int r, int c, float v) const {
    float y = fmaxf(v * sc[r] + bi[r], 0.0f);
    of[(size_t)r * ld + c] = y; oh[(size_t)r * ld + c] = (half_t)y;
  }
};
struct EpiHeadOut {  // optional bias, f16 store
  half_t* out; int ld; const float* bias;
  __device__ void operator()(int r, int c, float v) const {
    out[(size_t)r * ld + c] = (half_t)(bias ? (v + bias[r]) : v);
  }
};
struct EpiAttn {  // softmax numerator normalized by row stats
  half_t* attn; int ld; const float* rowmax; const float* rowinv;
  __device__ void operator()(int r, int c, float v) const {
    attn[(size_t)r * ld + c] = (half_t)(__expf(v - rowmax[r]) * rowinv[r]);
  }
};
struct EpiD {  // d = h - xr * colscale, f16 store
  half_t* d16; int ld; const float* hsrc; int ldh; const float* cs;
  __device__ void operator()(int r, int c, float v) const {
    d16[(size_t)r * ld + c] = (half_t)(hsrc[(size_t)r * ldh + c] - v * cs[c]);
  }
};
struct EpiTB {  // h' = h + relu(v*sc + bi), dual store
  float* hf; half_t* hh; int ld; const float* sc; const float* bi; const float* hold;
  __device__ void operator()(int r, int c, float v) const {
    float y = fmaxf(v * sc[r] + bi[r], 0.0f) + hold[(size_t)r * ld + c];
    hf[(size_t)r * ld + c] = y; hh[(size_t)r * ld + c] = (half_t)y;
  }
};
struct EpiProj {  // + bias, dual store
  float* hf; half_t* hh; int ld; const float* bias;
  __device__ void operator()(int r, int c, float v) const {
    float y = v + bias[r];
    hf[(size_t)r * ld + c] = y; hh[(size_t)r * ld + c] = (half_t)y;
  }
};

// ------------------------------ GEMM kernels -------------------------------
__global__ void k_gemm_in(const half_t* __restrict__ W, const half_t* __restrict__ Bm,
                          const float* sc, const float* bi, float* of, half_t* oh) {
  EpiSBRelu epi{of, oh, J_, sc, bi};
  gemm_core<false>(W, C_, Bm, J_, C_, epi);
}
__global__ void k_gemm_head_lin(const half_t* __restrict__ Wl, const half_t* __restrict__ h16,
                                const float* vb, half_t* outT) {
  int z = blockIdx.z, b = z >> 2, hh = z & 3;
  EpiHeadOut epi{outT + (size_t)z * HD_ * N_, N_, vb ? (vb + hh * HD_) : nullptr};
  gemm_core<false>(Wl + hh * HD_ * HD_, HD_, h16 + (size_t)hh * HD_ * J_ + b * N_, J_, HD_, epi);
}
__global__ void k_gemm_attn(const half_t* __restrict__ qT, const half_t* __restrict__ kT,
                            const float* rowmax, const float* rowinv, half_t* attn) {
  int z = blockIdx.z;
  EpiAttn epi{attn + (size_t)z * N_ * N_, N_, rowmax + z * N_, rowinv + z * N_};
  gemm_core<true>(qT + (size_t)z * HD_ * N_, N_, kT + (size_t)z * HD_ * N_, N_, HD_, epi);
}
__global__ void k_gemm_xr(const half_t* __restrict__ vT, const half_t* __restrict__ attn,
                          const float* cs, const float* hf, half_t* d16) {
  int z = blockIdx.z, b = z >> 2, hh = z & 3;
  EpiD epi{d16 + (size_t)z * HD_ * N_, N_, hf + (size_t)hh * HD_ * J_ + b * N_, J_, cs + z * N_};
  gemm_core<false>(vT + (size_t)z * HD_ * N_, N_, attn + (size_t)z * N_ * N_, N_, N_, epi);
}
__global__ void k_gemm_t(const half_t* __restrict__ Wl, const half_t* __restrict__ d16,
                         const float* sct, const float* bit, const float* hold,
                         float* hf, half_t* hh16) {
  int z = blockIdx.z, b = z >> 2, hh = z & 3;
  size_t co = (size_t)hh * HD_ * J_ + b * N_;
  EpiTB epi{hf + co, hh16 + co, J_, sct + hh * HD_, bit + hh * HD_, hold + co};
  gemm_core<false>(Wl + hh * HD_ * HD_, HD_, d16 + (size_t)z * HD_ * N_, N_, HD_, epi);
}
__global__ void k_gemm_proj(const half_t* __restrict__ Wl, const half_t* __restrict__ Bm,
                            const float* pb, float* hf, half_t* hh16) {
  EpiProj epi{hf, hh16, J_, pb};
  gemm_core<false>(Wl, C_, Bm, J_, C_, epi);
}

// --------------------------- softmax row stats -----------------------------
__global__ void k_stats(const half_t* __restrict__ qT, const half_t* __restrict__ kT,
                        float* rowmax, float* rowinv) {
  int z = blockIdx.y;
  int n = blockIdx.x * THREADS + threadIdx.x;
  const half_t* qz = qT + (size_t)z * HD_ * N_;
  const half_t* kz = kT + (size_t)z * HD_ * N_;
  float q[HD_];
#pragma unroll
  for (int c = 0; c < HD_; ++c) q[c] = (float)qz[(size_t)c * N_ + n];
  __shared__ half_t Ks[HD_][136];
  float mx = -1e30f, sum = 0.0f;
  for (int mt = 0; mt < N_; mt += 128) {
    __syncthreads();
#pragma unroll
    for (int it = 0; it < 4; ++it) {
      int idx = (threadIdx.x + it * THREADS) * 8;
      int c = idx >> 7, m = idx & 127;
#if HAVE_ASYNC
      ASYNC_CP_B128(kz + (size_t)c * N_ + mt + m, &Ks[c][m]);
#else
      *(v8h*)&Ks[c][m] = *(const v8h*)(kz + (size_t)c * N_ + mt + m);
#endif
    }
#if HAVE_ASYNC
    asm volatile("s_wait_asynccnt 0x0" ::: "memory");
#endif
    __syncthreads();
    for (int m = 0; m < 128; ++m) {
      float s = 0.0f;
#pragma unroll
      for (int c = 0; c < HD_; ++c) s += q[c] * (float)Ks[c][m];
      float nm = fmaxf(mx, s);
      sum = sum * __expf(mx - nm) + __expf(s - nm);
      mx = nm;
    }
  }
  rowmax[z * N_ + n] = mx;
  rowinv[z * N_ + n] = 1.0f / sum;
}

// --------------------- column-sum -> reciprocal scale ----------------------
__global__ void k_colscale(const half_t* __restrict__ attn, float* cs) {
  int z = blockIdx.y;
  int m = blockIdx.x * THREADS + threadIdx.x;
  const half_t* az = attn + (size_t)z * N_ * N_;
  float s = 0.0f;
  for (int n = 0; n < N_; ++n) s += (float)az[(size_t)n * N_ + m];
  cs[z * N_ + m] = 1.0f / (1e-9f + s);
}

// ------------------------------- small utils -------------------------------
__global__ void k_cvt(const float* __restrict__ src, half_t* __restrict__ dst, int n) {
  for (int i = blockIdx.x * blockDim.x + threadIdx.x; i < n; i += gridDim.x * blockDim.x)
    dst[i] = (half_t)src[i];
}
__global__ void k_prep(const float* b1, const float* g1, const float* be1,
                       const float* b2, const float* g2, const float* be2,
                       const float* tb, const float* bg, const float* bb,
                       float* sc1, float* bi1, float* sc2, float* bi2,
                       float* sct, float* bit) {
  int i = blockIdx.x * blockDim.x + threadIdx.x;
  const float rs = rsqrtf(1.0f + 1e-5f);
  if (i < C_) {
    float s = g1[i] * rs; sc1[i] = s; bi1[i] = b1[i] * s + be1[i];
    float t = g2[i] * rs; sc2[i] = t; bi2[i] = b2[i] * t + be2[i];
  }
  if (i < L_ * H_ * HD_) {
    float s = bg[i] * rs; sct[i] = s; bit[i] = tb[i] * s + bb[i];
  }
}
__global__ void k_xT(const float* __restrict__ x, half_t* __restrict__ xT) {
  int gid = blockIdx.x * THREADS + threadIdx.x;   // C_*J_ = 1M
  int c = gid >> 12, j = gid & (J_ - 1);
  xT[gid] = (half_t)x[(size_t)j * C_ + c];
}
__global__ void k_out(const float* __restrict__ h, float* __restrict__ out) {
  int gid = blockIdx.x * THREADS + threadIdx.x;   // 1M; gid = j*C + c
  int j = gid >> 8, c = gid & (C_ - 1);
  out[gid] = h[(size_t)c * J_ + j];
}

// ------------------------------- launcher ----------------------------------
extern "C" void kernel_launch(void* const* d_in, const int* in_sizes, int n_in,
                              void* d_out, int out_size, void* d_ws, size_t ws_size,
                              hipStream_t stream) {
  const float* x     = (const float*)d_in[0];
  const float* in_w1 = (const float*)d_in[1];
  const float* in_b1 = (const float*)d_in[2];
  const float* in_g1 = (const float*)d_in[3];
  const float* in_be1= (const float*)d_in[4];
  const float* in_w2 = (const float*)d_in[5];
  const float* in_b2 = (const float*)d_in[6];
  const float* in_g2 = (const float*)d_in[7];
  const float* in_be2= (const float*)d_in[8];
  const float* q_w   = (const float*)d_in[9];
  const float* k_w   = (const float*)d_in[10];
  const float* v_w   = (const float*)d_in[11];
  const float* v_b   = (const float*)d_in[12];
  const float* t_w   = (const float*)d_in[13];
  const float* t_b   = (const float*)d_in[14];
  const float* bn_g  = (const float*)d_in[15];
  const float* bn_b  = (const float*)d_in[16];
  const float* proj_w= (const float*)d_in[17];
  const float* proj_b= (const float*)d_in[18];
  float* out = (float*)d_out;

  uint8_t* base = (uint8_t*)d_ws;
  size_t off = 0;
  auto alloc = [&](size_t bytes) -> void* {
    void* p = base + off; off += (bytes + 255) & ~(size_t)255; return p;
  };
  float*  hA   = (float*) alloc((size_t)C_ * J_ * 4);
  float*  hB   = (float*) alloc((size_t)C_ * J_ * 4);
  half_t* hA16 = (half_t*)alloc((size_t)C_ * J_ * 2);
  half_t* hB16 = (half_t*)alloc((size_t)C_ * J_ * 2);
  half_t* xT16 = (half_t*)alloc((size_t)C_ * J_ * 2);
  half_t* w16  = (half_t*)alloc((size_t)655360 * 2);
  half_t* qT   = (half_t*)alloc((size_t)Z_ * HD_ * N_ * 2);
  half_t* kT   = (half_t*)alloc((size_t)Z_ * HD_ * N_ * 2);
  half_t* vT   = (half_t*)alloc((size_t)Z_ * HD_ * N_ * 2);
  half_t* dT   = (half_t*)alloc((size_t)Z_ * HD_ * N_ * 2);
  half_t* attn = (half_t*)alloc((size_t)Z_ * N_ * N_ * 2);
  float*  rmax = (float*) alloc((size_t)Z_ * N_ * 4);
  float*  rinv = (float*) alloc((size_t)Z_ * N_ * 4);
  float*  cs   = (float*) alloc((size_t)Z_ * N_ * 4);
  float*  sc1  = (float*) alloc(C_ * 4);
  float*  bi1  = (float*) alloc(C_ * 4);
  float*  sc2  = (float*) alloc(C_ * 4);
  float*  bi2  = (float*) alloc(C_ * 4);
  float*  sct  = (float*) alloc(L_ * H_ * HD_ * 4);
  float*  bit  = (float*) alloc(L_ * H_ * HD_ * 4);

  half_t* w_in1 = w16;
  half_t* w_in2 = w16 + 65536;
  half_t* w_q   = w16 + 131072;
  half_t* w_k   = w16 + 196608;
  half_t* w_v   = w16 + 262144;
  half_t* w_t   = w16 + 327680;
  half_t* w_p   = w16 + 393216;

  // weight conversion f32 -> f16
  k_cvt<<<256, THREADS, 0, stream>>>(in_w1,  w_in1, 65536);
  k_cvt<<<256, THREADS, 0, stream>>>(in_w2,  w_in2, 65536);
  k_cvt<<<256, THREADS, 0, stream>>>(q_w,    w_q,   65536);
  k_cvt<<<256, THREADS, 0, stream>>>(k_w,    w_k,   65536);
  k_cvt<<<256, THREADS, 0, stream>>>(v_w,    w_v,   65536);
  k_cvt<<<256, THREADS, 0, stream>>>(t_w,    w_t,   65536);
  k_cvt<<<256, THREADS, 0, stream>>>(proj_w, w_p,   262144);
  k_prep<<<4, THREADS, 0, stream>>>(in_b1, in_g1, in_be1, in_b2, in_g2, in_be2,
                                    t_b, bn_g, bn_b, sc1, bi1, sc2, bi2, sct, bit);
  k_xT<<<(C_ * J_) / THREADS, THREADS, 0, stream>>>(x, xT16);

  // input MLP (C x C GEMMs over 4096 columns)
  dim3 gC(J_ / BN, C_ / BM, 1);
  k_gemm_in<<<gC, THREADS, 0, stream>>>(w_in1, xT16, sc1, bi1, hB, hB16);
  k_gemm_in<<<gC, THREADS, 0, stream>>>(w_in2, hB16, sc2, bi2, hA, hA16);

  dim3 gH(N_ / BN, 1, Z_);
  dim3 gS(N_ / BN, N_ / BM, Z_);
  for (int l = 0; l < L_; ++l) {
    const half_t* qwl = w_q + (size_t)l * H_ * HD_ * HD_;
    const half_t* kwl = w_k + (size_t)l * H_ * HD_ * HD_;
    const half_t* vwl = w_v + (size_t)l * H_ * HD_ * HD_;
    const half_t* twl = w_t + (size_t)l * H_ * HD_ * HD_;
    const half_t* pwl = w_p + (size_t)l * C_ * C_;

    k_gemm_head_lin<<<gH, THREADS, 0, stream>>>(qwl, hA16, nullptr, qT);
    k_gemm_head_lin<<<gH, THREADS, 0, stream>>>(kwl, hA16, nullptr, kT);
    k_gemm_head_lin<<<gH, THREADS, 0, stream>>>(vwl, hA16, v_b + l * H_ * HD_, vT);

    k_stats<<<dim3(N_ / THREADS, Z_), THREADS, 0, stream>>>(qT, kT, rmax, rinv);
    k_gemm_attn<<<gS, THREADS, 0, stream>>>(qT, kT, rmax, rinv, attn);
    k_colscale<<<dim3(N_ / THREADS, Z_), THREADS, 0, stream>>>(attn, cs);
    k_gemm_xr<<<gH, THREADS, 0, stream>>>(vT, attn, cs, hA, dT);
    k_gemm_t<<<gH, THREADS, 0, stream>>>(twl, dT, sct + l * H_ * HD_, bit + l * H_ * HD_,
                                         hA, hB, hB16);
    k_gemm_proj<<<gC, THREADS, 0, stream>>>(pwl, hB16, proj_b + l * C_, hA, hA16);
  }
  k_out<<<(C_ * J_) / THREADS, THREADS, 0, stream>>>(hA, out);
}
